// EncoderSO3_37374805410388
// MI455X (gfx1250) — matrix operand order repaired and checked by reading, not compiled, not run on hardware
//
#include <hip/hip_runtime.h>
#include <cstdint>

typedef __attribute__((ext_vector_type(16))) _Float16 v16h;
typedef __attribute__((ext_vector_type(8)))  float    v8f;

// ===================== compile-time problem structure =====================
constexpr int   LMX    = 7;
constexpr int   BATCH  = 2048;
constexpr int   NB     = 4;      // batch elements per workgroup
constexpr int   NT     = 256;    // threads per workgroup (8 waves of 32)
constexpr int   NWAVES = NT / 32;
constexpr float EPS_   = 1e-5f;
constexpr int   NCPAD  = 240;    // padded N for stage-1 tiles (max NC = 15*15 = 225)
constexpr int   CINMAX = 384;    // max 16*N_PATHS[l]

struct PathTables {
    int n;
    short l1[160], l2[160], l3[160], slot[160];
    int npaths[LMX + 1];
};
constexpr PathTables make_paths() {
    PathTables t{};
    for (int a = 0; a <= LMX; ++a)
        for (int b = a; b <= LMX; ++b) {
            int hi = (a + b < LMX) ? (a + b) : LMX;
            for (int c = b - a; c <= hi; ++c) {
                t.l1[t.n] = (short)a; t.l2[t.n] = (short)b; t.l3[t.n] = (short)c;
                t.slot[t.n] = (short)t.npaths[c];
                t.npaths[c] += 1;
                t.n += 1;
            }
        }
    return t;
}
constexpr PathTables HPT = make_paths();
constexpr int NPATH = HPT.n;
static_assert(NPATH == 154, "unexpected path count");

struct Derived {
    int cin[LMX + 1];
    int hbase[LMX + 1];   // per-degree base offset into LDS h buffer (halves)
    int htot;             // halves total (includes NB factor)
};
constexpr Derived make_derived() {
    Derived d{};
    int off = 0;
    for (int l = 0; l <= LMX; ++l) {
        d.cin[l]   = 16 * HPT.npaths[l];
        d.hbase[l] = off;
        off += d.cin[l] * (2 * l + 1) * NB;
    }
    d.htot = off;
    return d;
}
constexpr Derived HDV = make_derived();

__constant__ PathTables cPT = HPT;
__constant__ Derived    cDV = HDV;

// ===================== LDS layout (dynamic shared, all offsets 16B aligned) =====================
constexpr int OFF_H    = 0;                               // _Float16[htot]      block hidden state
constexpr int OFF_N    = OFF_H    + HDV.htot * 2;         // float[NB*CINMAX]    per-(b,c) norms
constexpr int OFF_SCLH = OFF_N    + NB * CINMAX * 4;      // _Float16[NB*CINMAX] gate/sigma scale (zero-padded)
constexpr int OFF_HID  = OFF_SCLH + NB * CINMAX * 2;      // float[NB*4*64]      hid (f32, for VALU stage 2)
constexpr int OFF_HIDH = OFF_HID  + NB * 4 * 64 * 4;      // _Float16[NB*4*64]   hid shadow (f16, WMMA A)
constexpr int OFF_WF   = OFF_HIDH + NB * 4 * 64 * 2;      // _Float16[CINMAX*4]  staged W (zero-padded)
constexpr int OFF_ONES = OFF_WF   + CINMAX * 4 * 2;       // _Float16[512]       ones (unifies l==0 A pack)
constexpr int OFF_WB   = OFF_ONES + 512 * 2;              // float[3968]         raw w3j tile (async copy dst)
constexpr int OFF_TMP  = OFF_WB   + 3968 * 4;             // float[16*NCPAD]     stage-1 WMMA output
constexpr int OFF_ACC  = OFF_TMP  + 16 * NCPAD * 4;       // float[NB*64]        final channel sums
constexpr int OFF_RED  = OFF_ACC  + NB * 64 * 4;          // float[64]           tiny reduction scratch
constexpr int SMEM_BYTES = OFF_RED + 256;                 // ~228 KB < 320 KB WGP LDS

struct KArgs {
    const float* x[8];
    const float* w3j[NPATH];
    const float* gamma[3][8];
    const float* beta[3][8];
    const float* W[3][8];
    float* out;
};

__device__ __forceinline__ float swishf(float x) { return x / (1.0f + __expf(-x)); }

__device__ __forceinline__ v8f wmma16(v16h a, v16h b, v8f c) {
    // v_wmma_f32_16x16x32_f16 (f16 A/B, f32 accumulate)
    return __builtin_amdgcn_wmma_f32_16x16x32_f16(false, a, false, b, (short)0, c, false, false);
}

// ===================== fused 3-block SO(3) encoder =====================
__global__ void __launch_bounds__(NT, 1) enc_so3_fused(KArgs args) {
    extern __shared__ char smem[];
    _Float16* sh_h    = (_Float16*)(smem + OFF_H);
    float*    sh_n    = (float*)   (smem + OFF_N);
    _Float16* sh_sclh = (_Float16*)(smem + OFF_SCLH);
    float*    sh_hid  = (float*)   (smem + OFF_HID);
    _Float16* sh_hidh = (_Float16*)(smem + OFF_HIDH);
    _Float16* sh_wfh  = (_Float16*)(smem + OFF_WF);
    _Float16* sh_ones = (_Float16*)(smem + OFF_ONES);
    float*    sh_wb   = (float*)   (smem + OFF_WB);
    float*    sh_tmp  = (float*)   (smem + OFF_TMP);
    float*    sh_acc  = (float*)   (smem + OFF_ACC);
    float*    sh_red  = (float*)   (smem + OFF_RED);

    const int tid  = threadIdx.x;
    const int lane = tid & 31;
    const int wid  = tid >> 5;
    const int g    = lane >> 4;   // lane group (ISA register layouts)
    const int ln16 = lane & 15;
    const int b0   = blockIdx.x * NB;

    for (int i = tid; i < 512; i += NT) sh_ones[i] = (_Float16)1.0f;
    __syncthreads();

    for (int blk = 0; blk < 3; ++blk) {
        // ---------- Phase 1: build hid[b][d][l^2+m] (f32 + f16 shadow) ----------
        if (blk == 0) {
            for (int l = 0; l <= LMX; ++l) {
                const int dim = 2 * l + 1;
                const float gam = args.gamma[0][l][0];
                const float bet = args.beta[0][l][0];
                const float* W0 = args.W[0][l];
                for (int idx = tid; idx < NB * dim; idx += NT) {
                    const int b = idx / dim, m = idx % dim;
                    const float xv = args.x[l][(size_t)(b0 + b) * dim + m];
                    float tval;
                    if (l == 0) {
                        tval = swishf(bet);    // (x-mu)==0 when cin==1
                    } else {
                        float s2 = 0.f;
                        for (int mm = 0; mm < dim; ++mm) {
                            float xx = args.x[l][(size_t)(b0 + b) * dim + mm];
                            s2 += xx * xx;
                        }
                        const float n    = sqrtf(s2 / dim + EPS_);
                        const float sig  = sqrtf(n * n + EPS_);
                        const float nh   = n / sig;
                        const float gate = swishf(gam * nh + bet) / (nh + EPS_);
                        tval = (xv / sig) * gate;
                    }
                    for (int d = 0; d < 4; ++d) {
                        const float hv = tval * W0[d];
                        sh_hid [(b * 4 + d) * 64 + l * l + m] = hv;
                        sh_hidh[(b * 4 + d) * 64 + l * l + m] = (_Float16)hv;
                    }
                }
            }
            __syncthreads();
        } else {
            for (int l = 0; l <= LMX; ++l) {
                const int dim  = 2 * l + 1;
                const int cin  = cDV.cin[l];
                const int CINP = (cin + 31) & ~31;
                const _Float16* hl = sh_h + cDV.hbase[l];
                const float* gam = args.gamma[blk][l];
                const float* bet = args.beta[blk][l];
                const float* Wl  = args.W[blk][l];
                // stage W into zero-padded f16 LDS
                for (int i = tid; i < CINP * 4; i += NT) {
                    const int c = i >> 2;
                    sh_wfh[i] = (c < cin) ? (_Float16)Wl[i] : (_Float16)0;
                }
                if (l == 0) {
                    if (tid < NB) {
                        float mu = 0.f;
                        for (int c = 0; c < cin; ++c) mu += (float)hl[c * NB + tid];
                        mu /= cin;
                        float var = 0.f;
                        for (int c = 0; c < cin; ++c) {
                            const float d0 = (float)hl[c * NB + tid] - mu; var += d0 * d0;
                        }
                        var /= cin;
                        sh_red[tid * 2 + 0] = mu;
                        sh_red[tid * 2 + 1] = 1.0f / sqrtf(var + EPS_);
                    }
                    __syncthreads();
                    for (int idx = tid; idx < NB * CINP; idx += NT) {
                        const int b = idx % NB, c = idx / NB;
                        _Float16 o = (_Float16)0;
                        if (c < cin) {
                            const float xv = (float)hl[c * NB + b];
                            o = (_Float16)swishf(gam[c] * (xv - sh_red[b * 2]) * sh_red[b * 2 + 1] + bet[c]);
                        }
                        sh_sclh[c * NB + b] = o;
                    }
                    __syncthreads();
                } else {
                    for (int idx = tid; idx < NB * cin; idx += NT) {
                        const int b = idx % NB, c = idx / NB;
                        float s2 = 0.f;
                        const _Float16* p = hl + (size_t)c * dim * NB + b;
                        for (int k = 0; k < dim; ++k) { const float v = (float)p[k * NB]; s2 += v * v; }
                        sh_n[c * NB + b] = sqrtf(s2 / dim + EPS_);
                    }
                    __syncthreads();
                    if (tid < NB) {
                        float s = 0.f;
                        for (int c = 0; c < cin; ++c) { const float n = sh_n[c * NB + tid]; s += n * n; }
                        sh_red[tid] = sqrtf(s / cin + EPS_);   // sigma
                    }
                    __syncthreads();
                    for (int idx = tid; idx < NB * CINP; idx += NT) {
                        const int b = idx % NB, c = idx / NB;
                        _Float16 o = (_Float16)0;
                        if (c < cin) {
                            const float sig  = sh_red[b];
                            const float nh   = sh_n[c * NB + b] / sig;
                            const float gate = swishf(gam[c] * nh + bet[c]) / (nh + EPS_);
                            o = (_Float16)(gate / sig);
                        }
                        sh_sclh[c * NB + b] = o;
                    }
                    __syncthreads();
                }
                // WMMA channel mix: rows=(b,m) R<=60, K=c (padded to 32), N=16 (d<4 used)
                const int R   = NB * dim;
                const int ntm = (R + 15) / 16;
                const int kch = CINP / 32;
                const float isc = 1.0f / sqrtf((float)cin);
                const _Float16* hA = (l == 0) ? sh_ones : hl;   // ones-vector unifies l==0
                for (int tm = wid; tm < ntm; tm += NWAVES) {    // wave-uniform
                    const int  row   = tm * 16 + ln16;
                    const bool rowok = row < R;
                    const int  rowc  = rowok ? row : (R - 1);
                    const int  bb = rowc / dim, mm = rowc % dim;
                    const bool dok = ln16 < 4;
                    v8f cacc = {};
                    for (int kc = 0; kc < kch; ++kc) {
                        v16h av, bv;
#pragma unroll
                        for (int v = 0; v < 8; ++v) {
                            const int KA  = ((v & 4) ? 16 : 0) + (v & 3) * 2 + g * 8; // A K-slot layout
                            const int KB  = ((v & 4) ? 16 : 0) + (v & 3) * 2 + g * 8; // B K-slot layout
                            const int cA0 = kc * 32 + KA, cA1 = cA0 + 1;
                            const int cB0 = kc * 32 + KB;
                            const int cH0 = (cA0 < cin) ? cA0 : (cin - 1);  // clamp: no NaN garbage into WMMA
                            const int cH1 = (cA1 < cin) ? cA1 : (cin - 1);
                            const _Float16 p0 = hA[((size_t)cH0 * dim + mm) * NB + bb] * sh_sclh[cA0 * NB + bb];
                            const _Float16 p1 = hA[((size_t)cH1 * dim + mm) * NB + bb] * sh_sclh[cA1 * NB + bb];
                            av[2 * v]     = rowok ? p0 : (_Float16)0;
                            av[2 * v + 1] = rowok ? p1 : (_Float16)0;
                            const _Float16 w0 = sh_wfh[cB0 * 4 + (ln16 & 3)];
                            const _Float16 w1 = sh_wfh[(cB0 + 1) * 4 + (ln16 & 3)];
                            bv[2 * v]     = dok ? w0 : (_Float16)0;
                            bv[2 * v + 1] = dok ? w1 : (_Float16)0;
                        }
                        cacc = wmma16(av, bv, cacc);
                    }
#pragma unroll
                    for (int r = 0; r < 8; ++r) {               // D: M=r+8g, N=ln16
                        const int grow = tm * 16 + r + 8 * g;
                        if (grow < R && ln16 < 4) {
                            const int b2 = grow / dim, m2 = grow % dim;
                            const float val = cacc[r] * isc;
                            sh_hid [(b2 * 4 + ln16) * 64 + l * l + m2] = val;
                            sh_hidh[(b2 * 4 + ln16) * 64 + l * l + m2] = (_Float16)val;
                        }
                    }
                }
                __syncthreads();
            }
        }

        if (blk == 2) {
            for (int i = tid; i < NB * 64; i += NT) sh_acc[i] = 0.f;
            __syncthreads();
        }

        // ---------- Phase 2: Clebsch-Gordan paths ----------
        for (int p = 0; p < NPATH; ++p) {
            const int l1 = cPT.l1[p], l2 = cPT.l2[p], l3 = cPT.l3[p];
            const int d1 = 2 * l1 + 1, d2 = 2 * l2 + 1, d3 = 2 * l3 + 1;
            const int NC = d1 * d3;
            const float* w = args.w3j[p];
            // async DMA of the raw w3j tile into LDS (CDNA5 async path, ASYNCcnt tracked)
            {
                const int nel = d1 * d2 * d3;
                const int n4  = nel >> 2;
                for (int i = tid; i < n4; i += NT) {
                    const unsigned loff = (unsigned)OFF_WB + (unsigned)(i * 16);
                    const unsigned long long ga = (unsigned long long)(uintptr_t)(w + i * 4);
                    asm volatile("global_load_async_to_lds_b128 %0, %1, off" :: "v"(loff), "v"(ga) : "memory");
                }
                for (int i = (n4 << 2) + tid; i < nel; i += NT) {
                    const unsigned loff = (unsigned)OFF_WB + (unsigned)(i * 4);
                    const unsigned long long ga = (unsigned long long)(uintptr_t)(w + i);
                    asm volatile("global_load_async_to_lds_b32 %0, %1, off" :: "v"(loff), "v"(ga) : "memory");
                }
                asm volatile("s_wait_asynccnt 0x0" ::: "memory");
            }
            __syncthreads();
            // stage 1 (WMMA, M=16 exactly): tmp[(b,j),(m,k)] = sum_n hid_l2[b,j,n]*w3j[n,(m,k)]
            v16h av;
            {
                const int bb = ln16 >> 2, jj = ln16 & 3;   // row = b*4 + j
                const int hb = (bb * 4 + jj) * 64 + l2 * l2;
#pragma unroll
                for (int v = 0; v < 8; ++v) {
                    const int KA = ((v & 4) ? 16 : 0) + (v & 3) * 2 + g * 8;
                    const int k0 = (KA < d2) ? KA : (d2 - 1);
                    const int k1 = (KA + 1 < d2) ? (KA + 1) : (d2 - 1);
                    const _Float16 a0 = sh_hidh[hb + k0];
                    const _Float16 a1 = sh_hidh[hb + k1];
                    av[2 * v]     = (KA < d2)     ? a0 : (_Float16)0;
                    av[2 * v + 1] = (KA + 1 < d2) ? a1 : (_Float16)0;
                }
            }
            const int ntiles = (NC + 15) / 16;
            for (int nt = wid; nt < ntiles; nt += NWAVES) {    // wave-uniform
                const int col  = nt * 16 + ln16;
                const int colc = (col < NC) ? col : (NC - 1);
                const int m    = colc / d3;
                const int k    = colc - m * d3;
                const int base = m * d2 * d3 + k;              // raw [m][n][k]: + n*d3
                v16h bv;
#pragma unroll
                for (int v = 0; v < 8; ++v) {
                    const int KB = ((v & 4) ? 16 : 0) + (v & 3) * 2 + g * 8;
                    const float w0 = sh_wb[base + KB * d3];
                    const float w1 = sh_wb[base + (KB + 1) * d3];
                    bv[2 * v]     = (KB < d2)     ? (_Float16)w0 : (_Float16)0;
                    bv[2 * v + 1] = (KB + 1 < d2) ? (_Float16)w1 : (_Float16)0;
                }
                v8f cc = {};
                cc = wmma16(av, bv, cc);
#pragma unroll
                for (int r = 0; r < 8; ++r)
                    sh_tmp[(r + 8 * g) * NCPAD + col] = cc[r];
            }
            __syncthreads();
            // stage 2 (VALU): out[b,i,j,k] = sum_m hid_l1[b,i,m] * tmp[(b,j),(m,k)]
            if (blk < 2) {
                _Float16* hdst = sh_h + cDV.hbase[l3];
                const int cbase = 16 * cPT.slot[p];
                const int tot = NB * 16 * d3;
                for (int idx = tid; idx < tot; idx += NT) {
                    const int k  = idx % d3;
                    const int ij = (idx / d3) % 16;
                    const int b  = idx / (16 * d3);
                    const int i  = ij >> 2, j = ij & 3;
                    float s = 0.f;
                    for (int m = 0; m < d1; ++m)
                        s += sh_hid[(b * 4 + i) * 64 + l1 * l1 + m] * sh_tmp[(b * 4 + j) * NCPAD + m * d3 + k];
                    hdst[((size_t)(cbase + ij) * d3 + k) * NB + b] = (_Float16)s;
                }
            } else {
                const int tot = NB * d3;
                for (int idx = tid; idx < tot; idx += NT) {
                    const int k = idx % d3, b = idx / d3;
                    float s = 0.f;
                    for (int ij = 0; ij < 16; ++ij) {
                        const int i = ij >> 2, j = ij & 3;
                        for (int m = 0; m < d1; ++m)
                            s += sh_hid[(b * 4 + i) * 64 + l1 * l1 + m] * sh_tmp[(b * 4 + j) * NCPAD + m * d3 + k];
                    }
                    sh_acc[b * 64 + l3 * l3 + k] += s;   // single writer per (b,l3,k) per path
                }
            }
            __syncthreads();
        }
    }

    // ---------- write output: tensor l lives at flat offset BATCH*l^2 ----------
    for (int idx = tid; idx < NB * 64; idx += NT) {
        const int pos = idx % 64, b = idx / 64;
        int l = 0;
        while ((l + 1) * (l + 1) <= pos) ++l;
        const int k   = pos - l * l;
        const int dim = 2 * l + 1;
        const float val = sh_acc[b * 64 + pos] / (float)(16 * cPT.npaths[l]);
        args.out[(size_t)BATCH * l * l + (size_t)(b0 + b) * dim + k] = val;
    }
}

// ===================== host launcher =====================
extern "C" void kernel_launch(void* const* d_in, const int* in_sizes, int n_in,
                              void* d_out, int out_size, void* d_ws, size_t ws_size,
                              hipStream_t stream) {
    (void)in_sizes; (void)n_in; (void)d_ws; (void)ws_size; (void)out_size;
    KArgs a;
    int idx = 0;
    for (int l = 0; l < 8; ++l) a.x[l] = (const float*)d_in[idx++];
    for (int p = 0; p < NPATH; ++p) a.w3j[p] = (const float*)d_in[idx++];
    for (int b = 0; b < 3; ++b) {
        for (int l = 0; l < 8; ++l) a.gamma[b][l] = (const float*)d_in[idx++];
        for (int l = 0; l < 8; ++l) a.beta[b][l]  = (const float*)d_in[idx++];
        for (int l = 0; l < 8; ++l) a.W[b][l]     = (const float*)d_in[idx++];
    }
    a.out = (float*)d_out;

    (void)hipFuncSetAttribute(reinterpret_cast<const void*>(enc_so3_fused),
                              hipFuncAttributeMaxDynamicSharedMemorySize, SMEM_BYTES);
    enc_so3_fused<<<BATCH / NB, NT, SMEM_BYTES, stream>>>(a);
}
